// DenseRelationDistill_15874199126327
// MI455X (gfx1250) — matrix operand order repaired and checked by reading, not compiled, not run on hardware
//
#include <hip/hip_runtime.h>

typedef __attribute__((ext_vector_type(16))) __bf16 v16bf;
typedef __attribute__((ext_vector_type(8)))  __bf16 v8bf;
typedef __attribute__((ext_vector_type(8)))  float  v8f;

#define DEVINL static __device__ __forceinline__

DEVINL v8f wmma_bf16(v16bf a, v16bf b, v8f c) {
  // D = A(16x32 bf16) * B(32x16 bf16) + C(16x16 f32)
  return __builtin_amdgcn_wmma_f32_16x16x32_bf16(false, a, false, b, (short)0, c, false, false);
}
DEVINL __bf16 f2bf(float x) { return (__bf16)x; }
DEVINL int imin(int a, int b) { return a < b ? a : b; }

DEVINL v16bf pack16(v8bf lo, v8bf hi) {
  v16bf a;
#pragma unroll
  for (int i = 0; i < 8; ++i) { a[i] = lo[i]; a[8 + i] = hi[i]; }
  return a;
}

// ---------------------------------------------------------------------------
// Pad a (nimg,256,32,32) f32 NCHW tensor into zero-haloed NHWC bf16 (nimg,34,34,256).
__global__ void k_pad32(const float* __restrict__ in, __bf16* __restrict__ out, int total) {
  int gid = blockIdx.x * 256 + threadIdx.x;
  if (gid >= total) return;
  int c = gid & 255;
  int t = gid >> 8;
  int x = t % 34;
  t /= 34;
  int y = t % 34;
  int img = t / 34;
  __bf16 v = (__bf16)0.0f;
  if (y >= 1 && y <= 32 && x >= 1 && x <= 32)
    v = f2bf(in[(size_t)(img * 256 + c) * 1024 + (y - 1) * 32 + (x - 1)]);
  out[gid] = v;
}

// Per-(img,channel) spatial mean; one block per (b,c).
__global__ void k_mean(const float* __restrict__ f, float* __restrict__ mean, int S) {
  int bc = blockIdx.x;
  const float* src = f + (size_t)bc * S * S;
  float s = 0.f;
  for (int i = threadIdx.x; i < S * S; i += 256) s += src[i];
  __shared__ float r[256];
  r[threadIdx.x] = s;
  __syncthreads();
  for (int st = 128; st > 0; st >>= 1) {
    if (threadIdx.x < st) r[threadIdx.x] += r[threadIdx.x + st];
    __syncthreads();
  }
  if (threadIdx.x == 0) mean[bc] = r[0] / (float)(S * S);
}

// Align-corners bilinear resize (S->32) of centered feats -> padded NHWC bf16 (2,34,34,256).
__global__ void k_rcp(const float* __restrict__ f, const float* __restrict__ mean,
                      __bf16* __restrict__ out, int S, int total) {
  int gid = blockIdx.x * 256 + threadIdx.x;
  if (gid >= total) return;
  int c = gid & 255;
  int t = gid >> 8;
  int x = t % 34;
  t /= 34;
  int y = t % 34;
  int b = t / 34;
  __bf16 v = (__bf16)0.0f;
  if (y >= 1 && y <= 32 && x >= 1 && x <= 32) {
    int yy = y - 1, xx = x - 1;
    float scale = (float)(S - 1) / 31.0f;
    float fy = yy * scale, fx = xx * scale;
    int y0 = (int)fy, x0 = (int)fx;
    int y1 = imin(y0 + 1, S - 1), x1 = imin(x0 + 1, S - 1);
    float wy = fy - (float)y0, wx = fx - (float)x0;
    const float* src = f + (size_t)(b * 256 + c) * S * S;
    float v00 = src[y0 * S + x0], v01 = src[y0 * S + x1];
    float v10 = src[y1 * S + x0], v11 = src[y1 * S + x1];
    float r0v = v00 + (v01 - v00) * wx;
    float r1v = v10 + (v11 - v10) * wx;
    v = f2bf(r0v + (r1v - r0v) * wy - mean[b * 256 + c]);
  }
  out[gid] = v;
}

// Conv weights (O,256,3,3) f32 -> fragment-friendly [tap(9)][kc(8)][o(O)][32] bf16.
__global__ void k_wprep3(const float* __restrict__ w, __bf16* __restrict__ wt, int O, int total) {
  int gid = blockIdx.x * 256 + threadIdx.x;
  if (gid >= total) return;
  int k = gid & 31;
  int t5 = gid >> 5;
  int o = t5 % O;
  int t6 = t5 / O;
  int kc = t6 & 7;
  int tap = t6 >> 3;
  wt[gid] = f2bf(w[((size_t)o * 256 + kc * 32 + k) * 9 + tap]);
}

// Combine conv weights (256,512) f32 -> [kc(16)][o(256)][32] bf16.
__global__ void k_wprep1(const float* __restrict__ w, __bf16* __restrict__ wt) {
  int gid = blockIdx.x * 256 + threadIdx.x;  // 131072 total
  int k = gid & 31;
  int o = (gid >> 5) & 255;
  int kc = gid >> 13;
  wt[gid] = f2bf(w[(size_t)o * 512 + kc * 32 + k]);
}

__global__ void k_zero(float* __restrict__ p, int n) {
  int i = blockIdx.x * blockDim.x + threadIdx.x;
  if (i < n) p[i] = 0.f;
}

// ---------------------------------------------------------------------------
// 3x3 SAME conv as 9-tap implicit GEMM with WMMA. NHWC input: B fragments are
// single 32B contiguous loads per lane.
// inpad: (nimg,34,34,256) bf16; wt: [9][8][O][32] bf16.
// out: tr_out ? (nimg,1024,O) : (nimg,O,1024) bf16.
// 128 threads = 4 waves; each wave computes one 16o x 16x tile at one y row.
__global__ void __launch_bounds__(128) k_conv3(const __bf16* __restrict__ inpad,
                                               const __bf16* __restrict__ wt,
                                               const float* __restrict__ bias,
                                               __bf16* __restrict__ outbf, int O, int nimg,
                                               int tr_out) {
  int lane = threadIdx.x & 31;
  int wv = threadIdx.x >> 5;
  int g = lane >> 4, hl = lane & 15;
  int tile = blockIdx.x * 4 + wv;
  int xh = tile & 1;
  int y = (tile >> 1) & 31;
  int rest = tile >> 6;
  int otn = O >> 4;
  int ot = rest % otn;
  int img = rest / otn;
  if (img >= nimg) return;  // wave-uniform

  v8f acc = {};
  for (int tap = 0; tap < 9; ++tap) {
    int dy = tap / 3, dx = tap % 3;
    int row = y + dy;
    int col = xh * 16 + hl + dx;
    size_t base = ((size_t)(img * 34 + row) * 34 + col) * 256 + 16 * g;
    for (int kc = 0; kc < 8; ++kc) {
      const __bf16* wp = wt + (((size_t)(tap * 8 + kc) * O + ot * 16 + hl) << 5);
      v16bf a = pack16(*(const v8bf*)(wp + 16 * g), *(const v8bf*)(wp + 16 * g + 8));
      v16bf bfrag = *(const v16bf*)(inpad + base + kc * 32);
      acc = wmma_bf16(a, bfrag, acc);
    }
  }
  if (tr_out) {
    // [img][p][O]: 8 consecutive channels per lane -> one 16B store
    size_t ob = ((size_t)img * 1024 + y * 32 + xh * 16 + hl) * O + ot * 16 + 8 * g;
    v8bf sv;
#pragma unroll
    for (int r = 0; r < 8; ++r) sv[r] = f2bf(acc[r] + bias[ot * 16 + r + 8 * g]);
    *(v8bf*)(outbf + ob) = sv;
  } else {
#pragma unroll
    for (int r = 0; r < 8; ++r) {
      int o = ot * 16 + r + 8 * g;
      outbf[((size_t)img * O + o) * 1024 + y * 32 + xh * 16 + hl] = f2bf(acc[r] + bias[o]);
    }
  }
}

// ---------------------------------------------------------------------------
// Fused attention: per (b, t-tile of 16): loop n over 20 classes:
//   logits(1024q x 16t) = kq[b]^T * key_t[n]   (1 WMMA per 16q tile, K=C=32)
//   exact softmax over q (in LDS)
//   acc[v,t] += {vq[b] | val_t[n]}(v x q) * P(q x t)   (WMMA over q chunks)
// kq: (2,1024,32) bf16 (q-major -> straight async DMA into LDS).
// fin: (2,256,1024) f32, channels [0,128)=vq_sum, [128,256)=vt_sum.
__global__ void __launch_bounds__(256) k_attn(const __bf16* __restrict__ kq,
                                              const __bf16* __restrict__ keyt,
                                              const __bf16* __restrict__ vq,
                                              const __bf16* __restrict__ valt,
                                              float* __restrict__ fin) {
  extern __shared__ char smem[];
  __bf16* kqs = (__bf16*)smem;                 // [q][c] 1024*32 bf16 = 64 KB
  float* Ls = (float*)(smem + 65536);          // [q][t] 1024*16 f32 = 64 KB
  float* red = (float*)(smem + 131072);        // 256 f32
  float* colmax = (float*)(smem + 132096);     // 16 f32
  float* rcp = (float*)(smem + 132160);        // 16 f32
  __bf16* kts = (__bf16*)(smem + 132224);      // [c][t] 32*16 bf16

  int tid = threadIdx.x;
  int lane = tid & 31, wv = tid >> 5;
  int g = lane >> 4, hl = lane & 15;
  int b = blockIdx.x >> 6;
  int t0 = (blockIdx.x & 63) << 4;

  // stage kq[b] (already [q][c]) into LDS via async global->LDS DMA (ASYNCcnt path)
  {
    const __bf16* kqb = kq + (size_t)b * 1024 * 32;
    unsigned ldsbase = (unsigned)(size_t)(void*)kqs;  // LDS byte offset (flat low 32b)
    for (int i = tid; i < 4096; i += 256) {           // 4096 x 16B = 64 KB
      unsigned dst = ldsbase + i * 16;
      const __bf16* src = kqb + i * 8;
      asm volatile("global_load_async_to_lds_b128 %0, %1, off"
                   :: "v"(dst), "v"(src)
                   : "memory");
    }
    asm volatile("s_wait_asynccnt 0x0" ::: "memory");
  }

  v8f acc0 = {}, acc1 = {};
  int smt = tid & 15, seg = tid >> 4;  // softmax mapping
  __syncthreads();

  for (int n = 0; n < 20; ++n) {
    // key_t tile [c][t]
    for (int i = tid; i < 512; i += 256) {
      int c = i >> 4, tl = i & 15;
      kts[c * 16 + tl] = keyt[((size_t)n * 32 + c) * 1024 + t0 + tl];
    }
    __syncthreads();

    // logits: 64 q-tiles, 8 per wave, single WMMA each (K = 32)
    for (int i = 0; i < 8; ++i) {
      int qt = wv * 8 + i;
      const __bf16* ap = kqs + (qt * 16 + hl) * 32;
      v16bf a = pack16(*(const v8bf*)(ap + 16 * g), *(const v8bf*)(ap + 16 * g + 8));
      v16bf bfrag;
#pragma unroll
      for (int j = 0; j < 16; ++j) bfrag[j] = kts[(j + 16 * g) * 16 + hl];
      v8f z = {};
      z = wmma_bf16(a, bfrag, z);
#pragma unroll
      for (int r = 0; r < 8; ++r) Ls[(qt * 16 + r + 8 * g) * 16 + hl] = z[r];
    }
    __syncthreads();

    // softmax over q per t column (exact, two-pass)
    {
      float m = -3.4e38f;
      for (int q = seg * 64; q < seg * 64 + 64; ++q) m = fmaxf(m, Ls[q * 16 + smt]);
      red[seg * 16 + smt] = m;
      __syncthreads();
      if (tid < 16) {
        float mm = -3.4e38f;
        for (int sgi = 0; sgi < 16; ++sgi) mm = fmaxf(mm, red[sgi * 16 + tid]);
        colmax[tid] = mm;
      }
      __syncthreads();
      float cm = colmax[smt];
      float s = 0.f;
      for (int q = seg * 64; q < seg * 64 + 64; ++q) {
        float e = __expf(Ls[q * 16 + smt] - cm);
        Ls[q * 16 + smt] = e;
        s += e;
      }
      red[seg * 16 + smt] = s;
      __syncthreads();
      if (tid < 16) {
        float ss = 0.f;
        for (int sgi = 0; sgi < 16; ++sgi) ss += red[sgi * 16 + tid];
        rcp[tid] = 1.f / ss;
      }
      __syncthreads();
    }

    // prefetch next class's val_t panel while we crunch this one
    if (n + 1 < 20)
      __builtin_prefetch(valt + (size_t)(n + 1) * 128 * 1024 + tid * 512, 0, 1);

    // value aggregation: waves 0..3 use vq[b] (v 0..127), waves 4..7 use val_t[n]
    const __bf16* Abase = (wv < 4) ? (vq + (size_t)b * 128 * 1024)
                                   : (valt + (size_t)n * 128 * 1024);
    int r0 = (wv < 4) ? wv * 32 : (wv - 4) * 32;
    float rc = rcp[hl];
    for (int qc = 0; qc < 32; ++qc) {
      v16bf bfrag;
#pragma unroll
      for (int j = 0; j < 16; ++j)
        bfrag[j] = f2bf(Ls[(qc * 32 + j + 16 * g) * 16 + hl] * rc);
      const __bf16* ap0 = Abase + (size_t)(r0 + hl) * 1024 + qc * 32;
      v16bf a0 = pack16(*(const v8bf*)(ap0 + 16 * g), *(const v8bf*)(ap0 + 16 * g + 8));
      acc0 = wmma_bf16(a0, bfrag, acc0);
      const __bf16* ap1 = Abase + (size_t)(r0 + 16 + hl) * 1024 + qc * 32;
      v16bf a1 = pack16(*(const v8bf*)(ap1 + 16 * g), *(const v8bf*)(ap1 + 16 * g + 8));
      acc1 = wmma_bf16(a1, bfrag, acc1);
    }
    __syncthreads();
  }

#pragma unroll
  for (int r = 0; r < 8; ++r) {
    int v0 = wv * 32 + r + 8 * g;  // concat channel (vq_sum | vt_sum)
    fin[((size_t)b * 256 + v0) * 1024 + t0 + hl] = acc0[r];
    int v1 = wv * 32 + 16 + r + 8 * g;
    fin[((size_t)b * 256 + v1) * 1024 + t0 + hl] = acc1[r];
  }
}

// ---------------------------------------------------------------------------
// Per-channel sum/sumsq of the (virtually) resized fin over (2,H,H) via on-the-fly
// bilinear recompute; block-reduced atomics into stats[c*2 .. c*2+1].
__global__ void __launch_bounds__(256) k_stats(const float* __restrict__ fin,
                                               float* __restrict__ stats, int H) {
  int W = H;
  int chunks = (H * W + 4095) >> 12;
  int bc = blockIdx.x / chunks;
  int ch = blockIdx.x % chunks;
  int c = bc & 255;
  const float* src = fin + (size_t)bc * 1024;
  int p0 = ch << 12, p1 = imin(p0 + 4096, H * W);
  float scale = 31.0f / (float)(H - 1);
  float s = 0.f, ss = 0.f;
  for (int p = p0 + threadIdx.x; p < p1; p += 256) {
    int y = p / W, x = p - y * W;
    float fy = y * scale, fx = x * scale;
    int y0 = (int)fy, x0 = (int)fx;
    int y1 = imin(y0 + 1, 31), x1 = imin(x0 + 1, 31);
    float wy = fy - (float)y0, wx = fx - (float)x0;
    float v00 = src[y0 * 32 + x0], v01 = src[y0 * 32 + x1];
    float v10 = src[y1 * 32 + x0], v11 = src[y1 * 32 + x1];
    float r0v = v00 + (v01 - v00) * wx;
    float r1v = v10 + (v11 - v10) * wx;
    float v = r0v + (r1v - r0v) * wy;
    s += v;
    ss += v * v;
  }
  __shared__ float rs[256], rss[256];
  rs[threadIdx.x] = s;
  rss[threadIdx.x] = ss;
  __syncthreads();
  for (int st = 128; st > 0; st >>= 1) {
    if (threadIdx.x < st) {
      rs[threadIdx.x] += rs[threadIdx.x + st];
      rss[threadIdx.x] += rss[threadIdx.x + st];
    }
    __syncthreads();
  }
  if (threadIdx.x == 0) {
    atomicAdd(&stats[c * 2], rs[0]);
    atomicAdd(&stats[c * 2 + 1], rss[0]);
  }
}

// stats -> per-channel BN scale/shift
__global__ void k_bnprep(const float* __restrict__ stats, const float* __restrict__ gamma,
                         const float* __restrict__ beta, float* __restrict__ scsh, int HW) {
  int c = threadIdx.x;  // 256 threads
  float cnt = 2.0f * (float)HW;
  float mu = stats[c * 2] / cnt;
  float var = stats[c * 2 + 1] / cnt - mu * mu;
  float rstd = rsqrtf(var + 1e-5f);
  float scv = gamma[c] * rstd;
  scsh[c] = scv;
  scsh[256 + c] = beta[c] - mu * scv;
}

// Build X = concat(f, BN(resize(fin))) in GEMM-friendly [b][p][512] bf16;
// resize recomputed on the fly from L2-resident fin.
__global__ void k_xbuild(const float* __restrict__ f, const float* __restrict__ fin,
                         const float* __restrict__ scsh, __bf16* __restrict__ xbf,
                         int H, int total) {
  int gid = blockIdx.x * 256 + threadIdx.x;
  if (gid >= total) return;
  int c = gid & 511;
  int t = gid >> 9;
  int HW = H * H;
  int p = t % HW;
  int b = t / HW;
  float v;
  if (c < 256) {
    v = f[(size_t)(b * 256 + c) * HW + p];
  } else {
    int c2 = c - 256;
    int y = p / H, x = p - y * H;
    float scale = 31.0f / (float)(H - 1);
    float fy = y * scale, fx = x * scale;
    int y0 = (int)fy, x0 = (int)fx;
    int y1 = imin(y0 + 1, 31), x1 = imin(x0 + 1, 31);
    float wy = fy - (float)y0, wx = fx - (float)x0;
    const float* src = fin + (size_t)(b * 256 + c2) * 1024;
    float v00 = src[y0 * 32 + x0], v01 = src[y0 * 32 + x1];
    float v10 = src[y1 * 32 + x0], v11 = src[y1 * 32 + x1];
    float r0v = v00 + (v01 - v00) * wx;
    float r1v = v10 + (v11 - v10) * wx;
    v = (r0v + (r1v - r0v) * wy) * scsh[c2] + scsh[256 + c2];
  }
  xbf[gid] = f2bf(v);
}

// ---------------------------------------------------------------------------
// 1x1 combine conv as WMMA GEMM (M=256, K=512, N=HW): out = W * X + b.
// X in [b][p][512] bf16 -> contiguous 32B B-fragment loads.
__global__ void __launch_bounds__(128) k_comb(const __bf16* __restrict__ xbf,
                                              const __bf16* __restrict__ wtc,
                                              const float* __restrict__ combb,
                                              float* __restrict__ out, int HW) {
  int lane = threadIdx.x & 31;
  int wv = threadIdx.x >> 5;
  int g = lane >> 4, hl = lane & 15;
  int tilesp = HW >> 4;
  int tile = blockIdx.x * 4 + wv;
  int pt = tile % tilesp;
  int tmp = tile / tilesp;
  int ot = tmp & 15;
  int b = tmp >> 4;
  int p = pt * 16 + hl;
  const __bf16* xrow = xbf + ((size_t)b * HW + p) * 512 + 16 * g;

  v8f acc = {};
  for (int kc = 0; kc < 16; ++kc) {
    const __bf16* wp = wtc + (((size_t)kc * 256 + ot * 16 + hl) << 5);
    v16bf a = pack16(*(const v8bf*)(wp + 16 * g), *(const v8bf*)(wp + 16 * g + 8));
    v16bf bfrag = *(const v16bf*)(xrow + kc * 32);
    acc = wmma_bf16(a, bfrag, acc);
  }
#pragma unroll
  for (int r = 0; r < 8; ++r) {
    int o = ot * 16 + r + 8 * g;
    out[((size_t)b * 256 + o) * HW + p] = acc[r] + combb[o];
  }
}

// ---------------------------------------------------------------------------
extern "C" void kernel_launch(void* const* d_in, const int* in_sizes, int n_in,
                              void* d_out, int out_size, void* d_ws, size_t ws_size,
                              hipStream_t stream) {
  (void)in_sizes; (void)n_in; (void)out_size; (void)ws_size;
  const float* feat[5] = {(const float*)d_in[0], (const float*)d_in[1], (const float*)d_in[2],
                          (const float*)d_in[3], (const float*)d_in[4]};
  const float* attn     = (const float*)d_in[5];
  const float* key_t_w  = (const float*)d_in[6];
  const float* key_t_b  = (const float*)d_in[7];
  const float* val_t_w  = (const float*)d_in[8];
  const float* val_t_b  = (const float*)d_in[9];
  const float* key_q_w  = (const float*)d_in[10];
  const float* key_q_b  = (const float*)d_in[11];
  const float* val_q_w  = (const float*)d_in[12];
  const float* val_q_b  = (const float*)d_in[13];
  const float* bn_gamma = (const float*)d_in[14];
  const float* bn_beta  = (const float*)d_in[15];
  const float* comb_w   = (const float*)d_in[16];
  const float* comb_b   = (const float*)d_in[17];
  float* out = (float*)d_out;

  char* ws = (char*)d_ws;
  size_t off = 0;
  auto take = [&](size_t bytes) -> char* {
    char* p = ws + off;
    off += (bytes + 255) & ~(size_t)255;
    return p;
  };

  __bf16* attnpad = (__bf16*)take(20ull * 34 * 34 * 256 * 2);
  __bf16* frpad   = (__bf16*)take(2ull * 34 * 34 * 256 * 2);
  float*  meanbuf = (float*)take(512 * 4);
  __bf16* wt_keyt = (__bf16*)take(9ull * 8 * 32 * 32 * 2);
  __bf16* wt_valt = (__bf16*)take(9ull * 8 * 128 * 32 * 2);
  __bf16* wt_keyq = (__bf16*)take(5ull * 9 * 8 * 32 * 32 * 2);
  __bf16* wt_valq = (__bf16*)take(5ull * 9 * 8 * 128 * 32 * 2);
  __bf16* wt_comb = (__bf16*)take(16ull * 256 * 32 * 2);
  __bf16* keyt_bf = (__bf16*)take(20ull * 32 * 1024 * 2);
  __bf16* valt_bf = (__bf16*)take(20ull * 128 * 1024 * 2);
  __bf16* kq_bf   = (__bf16*)take(2ull * 1024 * 32 * 2);   // [b][q][c]
  __bf16* vq_bf   = (__bf16*)take(2ull * 128 * 1024 * 2);  // [b][v][q]
  float*  fin     = (float*)take(2ull * 256 * 1024 * 4);
  float*  stats   = (float*)take(512 * 4);
  float*  scsh    = (float*)take(512 * 4);
  __bf16* xbf     = (__bf16*)take(2ull * 16384 * 512 * 2);  // max level, reused

  // stage 0: pad attentions (NHWC), prep all weights
  {
    int total = 20 * 34 * 34 * 256;
    k_pad32<<<(total + 255) / 256, 256, 0, stream>>>(attn, attnpad, total);
  }
  {
    int tot = 32 * 2304;
    k_wprep3<<<(tot + 255) / 256, 256, 0, stream>>>(key_t_w, wt_keyt, 32, tot);
  }
  {
    int tot = 128 * 2304;
    k_wprep3<<<(tot + 255) / 256, 256, 0, stream>>>(val_t_w, wt_valt, 128, tot);
  }
  for (int l = 0; l < 5; ++l) {
    int tot = 32 * 2304;
    k_wprep3<<<(tot + 255) / 256, 256, 0, stream>>>(key_q_w + (size_t)l * 32 * 2304,
                                                    wt_keyq + (size_t)l * 9 * 8 * 32 * 32, 32, tot);
    tot = 128 * 2304;
    k_wprep3<<<(tot + 255) / 256, 256, 0, stream>>>(val_q_w + (size_t)l * 128 * 2304,
                                                    wt_valq + (size_t)l * 9 * 8 * 128 * 32, 128, tot);
  }
  k_wprep1<<<512, 256, 0, stream>>>(comb_w, wt_comb);

  // template projections (shared across levels): key_t [n][c][t], val_t [n][v][q]
  k_conv3<<<(20 * 2 * 64) / 4, 128, 0, stream>>>(attnpad, wt_keyt, key_t_b, keyt_bf, 32, 20, 0);
  k_conv3<<<(20 * 8 * 64) / 4, 128, 0, stream>>>(attnpad, wt_valt, val_t_b, valt_bf, 128, 20, 0);

  const size_t ATTN_LDS = 65536 + 65536 + 1024 + 64 + 64 + 1024;  // 133248 B of 320 KB WGP LDS
  static const int S[5] = {128, 64, 32, 16, 8};
  size_t ooff = 0;
  for (int l = 0; l < 5; ++l) {
    int s = S[l], HW = s * s;
    k_mean<<<512, 256, 0, stream>>>(feat[l], meanbuf, s);
    {
      int total = 2 * 34 * 34 * 256;
      k_rcp<<<(total + 255) / 256, 256, 0, stream>>>(feat[l], meanbuf, frpad, s, total);
    }
    // kq transposed to [b][q][c] for straight async-DMA LDS staging
    k_conv3<<<(2 * 2 * 64) / 4, 128, 0, stream>>>(frpad, wt_keyq + (size_t)l * 9 * 8 * 32 * 32,
                                                  key_q_b + l * 32, kq_bf, 32, 2, 1);
    k_conv3<<<(2 * 8 * 64) / 4, 128, 0, stream>>>(frpad, wt_valq + (size_t)l * 9 * 8 * 128 * 32,
                                                  val_q_b + l * 128, vq_bf, 128, 2, 0);
    k_attn<<<128, 256, ATTN_LDS, stream>>>(kq_bf, keyt_bf, vq_bf, valt_bf, fin);
    k_zero<<<2, 256, 0, stream>>>(stats, 512);
    int chunks = (HW + 4095) / 4096;
    k_stats<<<512 * chunks, 256, 0, stream>>>(fin, stats, s);
    k_bnprep<<<1, 256, 0, stream>>>(stats, bn_gamma + l * 256, bn_beta + l * 256, scsh, HW);
    {
      int total = 2 * HW * 512;
      k_xbuild<<<(total + 255) / 256, 256, 0, stream>>>(feat[l], fin, scsh, xbf, s, total);
    }
    int tilesp = HW / 16;
    k_comb<<<(2 * 16 * tilesp) / 4, 128, 0, stream>>>(xbf, wt_comb, comb_b, out + ooff, HW);
    ooff += (size_t)2 * 256 * HW;
  }
}
